// Linear4Bit_11991548690499
// MI455X (gfx1250) — matrix operand order repaired and checked by reading, not compiled, not run on hardware
//
#include <hip/hip_runtime.h>
#include <hip/hip_bf16.h>

typedef __attribute__((ext_vector_type(16))) __bf16 v16bf;
typedef __attribute__((ext_vector_type(8)))  __bf16 v8bf;
typedef __attribute__((ext_vector_type(4)))  __bf16 v4bf;
typedef __attribute__((ext_vector_type(8)))  float  v8f;

#define TOKENS 8192
#define IN_F   4096
#define OUT_F  4096
#define GROUPS (IN_F / 16)
#define PACKED (IN_F / 2)

#define BM 128
#define BN 128
#define BK 64
#define LDT 72              // padded bf16 row stride (144 B, 16B-aligned, conflict-free)
#define NITER (IN_F / BK)   // 64 iterations, 1 barrier each

__global__ __launch_bounds__(256) void linear4bit_wmma_bf16(
    const float* __restrict__ x,      // [TOKENS, IN_F] fp32
    const int*   __restrict__ wq,     // [OUT_F, PACKED] one byte per int, 2 nibbles
    const float* __restrict__ wn,     // [OUT_F, GROUPS] fp32 scales
    const float* __restrict__ bias,   // [OUT_F]
    float*       __restrict__ out)    // [TOKENS, OUT_F]
{
    __shared__ __bf16 As[2][BM * LDT];
    __shared__ __bf16 Bs[2][BN * LDT];

    const int tid  = threadIdx.x;
    const int lane = tid & 31;
    const int wave = tid >> 5;
    const int wm   = wave & 3;   // 4 wave-rows x 32 rows
    const int wn_  = wave >> 2;  // 2 wave-cols x 64 cols

    const int row0 = blockIdx.y * BM;
    const int col0 = blockIdx.x * BN;

    const int lm = lane & 15;
    const int lk = lane >> 4;

    v8f zero = {};
    v8f acc[2][4];
#pragma unroll
    for (int i = 0; i < 2; ++i)
#pragma unroll
        for (int j = 0; j < 4; ++j) acc[i][j] = zero;

    // ---- global load + eager convert to bf16 staging registers ----
    auto load_convert = [&](int kb, v4bf a4[8], v8bf b8[4]) {
#pragma unroll
        for (int i = 0; i < 8; ++i) {              // A: 128 rows x 16 float4
            int idx = tid + 256 * i;
            int r = idx >> 4, c4 = idx & 15;
            float4 t = *(const float4*)(x + (size_t)(row0 + r) * IN_F + kb + c4 * 4);
            v4bf a;
            a[0] = (__bf16)t.x; a[1] = (__bf16)t.y;
            a[2] = (__bf16)t.z; a[3] = (__bf16)t.w;
            a4[i] = a;
        }
#pragma unroll
        for (int i = 0; i < 4; ++i) {              // B: 128 rows x 8 int4
            int idx = tid + 256 * i;
            int r = idx >> 3, j4 = idx & 7;
            int4 q = *(const int4*)(wq + (size_t)(col0 + r) * PACKED + (kb >> 1) + j4 * 4);
            // 8 k-values of one int4 share one scale group of 16
            float n = wn[(size_t)(col0 + r) * GROUPS + ((kb + j4 * 8) >> 4)];
            float s = n * (2.0f / 15.0f);          // w = s*q - n
            int vals[4] = { q.x, q.y, q.z, q.w };
            v8bf b;
#pragma unroll
            for (int e = 0; e < 4; ++e) {
                int lo = vals[e] & 15;
                int hi = (vals[e] >> 4) & 15;
                b[2 * e + 0] = (__bf16)(s * (float)lo - n);
                b[2 * e + 1] = (__bf16)(s * (float)hi - n);
            }
            b8[i] = b;
        }
    };

    auto store_tile = [&](int buf, const v4bf a4[8], const v8bf b8[4]) {
#pragma unroll
        for (int i = 0; i < 8; ++i) {
            int idx = tid + 256 * i;
            int r = idx >> 4, c4 = idx & 15;
            *(v4bf*)(&As[buf][r * LDT + c4 * 4]) = a4[i];   // ds_store_b64
        }
#pragma unroll
        for (int i = 0; i < 4; ++i) {
            int idx = tid + 256 * i;
            int r = idx >> 3, j4 = idx & 7;
            *(v8bf*)(&Bs[buf][r * LDT + j4 * 8]) = b8[i];   // ds_store_b128
        }
    };

    auto compute = [&](int buf) {
#pragma unroll
        for (int ks = 0; ks < 2; ++ks) {           // two 16x16x32 k-substeps
            const int k0 = ks * 32;
            v16bf afrag[2];
#pragma unroll
            for (int i = 0; i < 2; ++i) {
                const __bf16* base = &As[buf][(wm * 32 + i * 16 + lm) * LDT + k0];
                v8bf lo = *(const v8bf*)(base + 8 * lk);        // K = 8*lk..+7
                v8bf hi = *(const v8bf*)(base + 16 + 8 * lk);   // K = 16+8*lk..+7
#pragma unroll
                for (int e = 0; e < 8; ++e) { afrag[i][e] = lo[e]; afrag[i][e + 8] = hi[e]; }
            }
            v16bf bfrag[4];
#pragma unroll
            for (int j = 0; j < 4; ++j) {
                const __bf16* base = &Bs[buf][(wn_ * 64 + j * 16 + lm) * LDT + k0];
                v8bf b0 = *(const v8bf*)(base + 8 * lk);
                v8bf b1 = *(const v8bf*)(base + 16 + 8 * lk);
#pragma unroll
                for (int e = 0; e < 8; ++e) { bfrag[j][e] = b0[e]; bfrag[j][e + 8] = b1[e]; }
            }
#pragma unroll
            for (int i = 0; i < 2; ++i)
#pragma unroll
                for (int j = 0; j < 4; ++j)
                    acc[i][j] = __builtin_amdgcn_wmma_f32_16x16x32_bf16(
                        false, afrag[i], false, bfrag[j],
                        (short)0, acc[i][j], false, false);
        }
    };

    // ---- prologue: fill buffer 0 ----
    {
        v4bf a4[8]; v8bf b8[4];
        load_convert(0, a4, b8);
        store_tile(0, a4, b8);
    }
    __syncthreads();

    // ---- pipelined main loop: 1 barrier per 64-K iteration ----
    for (int it = 0; it < NITER; ++it) {
        const int cur = it & 1;
        v4bf na4[8]; v8bf nb8[4];
        const bool has_next = (it + 1 < NITER);
        if (has_next) load_convert((it + 1) * BK, na4, nb8);  // in flight during WMMAs

        compute(cur);                                          // 16 v_wmma per wave

        if (has_next) {
            store_tile(cur ^ 1, na4, nb8);                     // other buffer: no pre-barrier
            __syncthreads();                                   // publish + protect cur
        }
    }

    // ---- epilogue: C layout (VGPR r -> row r + 8*lk, col = lm), add bias ----
#pragma unroll
    for (int i = 0; i < 2; ++i) {
#pragma unroll
        for (int j = 0; j < 4; ++j) {
            int gcol = col0 + wn_ * 64 + j * 16 + lm;
            float bv = bias[gcol];
            int rbase = row0 + wm * 32 + i * 16 + 8 * lk;
#pragma unroll
            for (int r = 0; r < 8; ++r)
                out[(size_t)(rbase + r) * OUT_F + gcol] = acc[i][j][r] + bv;
        }
    }
}

extern "C" void kernel_launch(void* const* d_in, const int* in_sizes, int n_in,
                              void* d_out, int out_size, void* d_ws, size_t ws_size,
                              hipStream_t stream) {
    (void)in_sizes; (void)n_in; (void)d_ws; (void)ws_size; (void)out_size;
    const float* x    = (const float*)d_in[0];
    const int*   wq   = (const int*)d_in[1];
    const float* wn   = (const float*)d_in[2];
    const float* bias = (const float*)d_in[3];
    float*       out  = (float*)d_out;

    dim3 grid(OUT_F / BN, TOKENS / BM);   // (32, 64)
    linear4bit_wmma_bf16<<<grid, 256, 0, stream>>>(x, wq, wn, bias, out);
}
